// CFGEncoder_12601434046916
// MI455X (gfx1250) — compile-verified, matrix operand order checked by reading, not run on hardware
//
#include <hip/hip_runtime.h>
#include <hip/hip_bf16.h>

#define N_NODES   100000
#define NUM_GRAPHS 64
#define OUT_DIM   128
#define BN_EPS    1e-5f
#define M_BLK     80            // rows per block: 100000 = 80 * 1250
#define M_TILES   5             // 5 WMMA M-tiles per wave

typedef __bf16 v16bf __attribute__((ext_vector_type(16)));
typedef float  v8f   __attribute__((ext_vector_type(8)));

static __device__ __forceinline__ unsigned int f2bf(float f) {
  unsigned int u = __float_as_uint(f);
  u += 0x7FFFu + ((u >> 16) & 1u);          // round-to-nearest-even
  return u >> 16;
}

// ---------------------------------------------------------------- utilities
__global__ void zero_f32(float* __restrict__ p, long long n) {
  long long i = (long long)blockIdx.x * blockDim.x + threadIdx.x;
  long long stride = (long long)gridDim.x * blockDim.x;
  for (; i < n; i += stride) p[i] = 0.0f;
}

// Wt[n][k3] = stacked [Ws;Wi;Wo](k3, n) as bf16 (transposed so B-fragment
// loads are contiguous along K -> global_load_b128)
__global__ void wprep(const float* __restrict__ Ws, const float* __restrict__ Wi,
                      const float* __restrict__ Wo, unsigned short* __restrict__ Wt,
                      int din) {
  int K3 = 3 * din;
  int idx = blockIdx.x * blockDim.x + threadIdx.x;
  if (idx >= OUT_DIM * K3) return;
  int n = idx / K3, k3 = idx % K3;
  float v;
  if (k3 < din)          v = Ws[k3 * OUT_DIM + n];
  else if (k3 < 2 * din) v = Wi[(k3 - din) * OUT_DIM + n];
  else                   v = Wo[(k3 - 2 * din) * OUT_DIM + n];
  Wt[idx] = (unsigned short)f2bf(v);
}

// ------------------------------------------------------- edge scatter (HBM bound)
__global__ void scatter_edges(const float* __restrict__ h,
                              const int* __restrict__ src,
                              const int* __restrict__ dst,
                              float* __restrict__ agg_in,
                              float* __restrict__ agg_out,
                              int E, int din) {
  int per = din >> 2;                               // float4 chunks per row
  long long idx = (long long)blockIdx.x * blockDim.x + threadIdx.x;
  if (idx >= (long long)E * per) return;
  int e = (int)(idx / per);
  int j = (int)(idx % per) * 4;
  int s = src[e], d = dst[e];
  const float4 xs = *(const float4*)(h + (long long)s * din + j);
  const float4 xd = *(const float4*)(h + (long long)d * din + j);
  float* ai = agg_in  + (long long)d * din + j;
  float* ao = agg_out + (long long)s * din + j;
  atomicAdd(ai + 0, xs.x); atomicAdd(ai + 1, xs.y);
  atomicAdd(ai + 2, xs.z); atomicAdd(ai + 3, xs.w);
  atomicAdd(ao + 0, xd.x); atomicAdd(ao + 1, xd.y);
  atomicAdd(ao + 2, xd.z); atomicAdd(ao + 3, xd.w);
}

// --------------------------------------------- pack [x | agg_in | agg_out] -> bf16
__global__ void pack_bf16(const float* __restrict__ h, const float* __restrict__ ain,
                          const float* __restrict__ aout, unsigned int* __restrict__ Xcat,
                          int din) {
  int K3 = 3 * din, Kp = K3 >> 1;
  long long idx = (long long)blockIdx.x * blockDim.x + threadIdx.x;
  if (idx >= (long long)N_NODES * Kp) return;
  int i = (int)(idx / Kp);
  int k = (int)(idx % Kp) * 2;
  const float* sp; int kk;
  if (k < din)          { sp = h    + (long long)i * din; kk = k; }
  else if (k < 2 * din) { sp = ain  + (long long)i * din; kk = k - din; }
  else                  { sp = aout + (long long)i * din; kk = k - 2 * din; }
  unsigned int lo = f2bf(sp[kk]), hi = f2bf(sp[kk + 1]);
  Xcat[idx] = lo | (hi << 16);
}

// ------------------------------------------------ WMMA GEMM + ReLU + BN statistics
// grid.x = N/80 = 1250 blocks. 8 waves/block, wave w owns output columns w*16..w*16+15
// for 5 stacked 16-row M-tiles. A tile (80 x K3, bf16) staged to LDS via
// global_load_async_to_lds_b128; B fragments loaded once per K-step and reused 5x.
__global__ __launch_bounds__(256) void gemm_wmma(
    const unsigned short* __restrict__ Xcat,
    const unsigned short* __restrict__ Wt,
    float* __restrict__ Y,
    float* __restrict__ bnsum, float* __restrict__ bnsumsq,
    int K3) {
  __shared__ __align__(16) unsigned short As[M_BLK * 384];

  const int tid  = threadIdx.x;
  const int wave = tid >> 5;
  const int lane = tid & 31;
  const int m    = lane & 15;                       // A row / B column within tile
  const int g    = lane >> 4;                       // K half-group
  const int n    = wave * 16 + m;                   // global output column (0..127)
  const unsigned short* brow = Wt + (long long)n * K3;

  // warm the weight stream while the async LDS fill is in flight
  __builtin_prefetch(brow, 0, 3);
  __builtin_prefetch(brow + (K3 >> 1), 0, 3);

  // ---- async stage of the contiguous 80 x K3 A-tile into LDS (ASYNCcnt path)
  {
    const unsigned short* gbase = Xcat + (long long)blockIdx.x * M_BLK * K3; // uniform
    const unsigned lds_base = (unsigned)(size_t)(&As[0]);
    const int nvec = (M_BLK * K3) >> 3;             // 16-byte packets
    for (int v = tid; v < nvec; v += 256) {
      unsigned byteoff = (unsigned)v << 4;
      asm volatile("global_load_async_to_lds_b128 %0, %1, %2"
                   :
                   : "v"(lds_base + byteoff), "v"(byteoff), "s"(gbase)
                   : "memory");
    }
    asm volatile("s_wait_asynccnt 0x0" ::: "memory");
  }
  __syncthreads();

  v8f acc[M_TILES];
#pragma unroll
  for (int mt = 0; mt < M_TILES; ++mt)
    acc[mt] = (v8f){0.f, 0.f, 0.f, 0.f, 0.f, 0.f, 0.f, 0.f};

  const int steps = K3 >> 5;                        // K=32 per WMMA
  for (int s = 0; s < steps; ++s) {
    const int kk = s * 32 + 8 * g;
    union { uint4 q[2]; v16bf v; } b;
    b.q[0] = *(const uint4*)(brow + kk);            // K = kk .. kk+7
    b.q[1] = *(const uint4*)(brow + kk + 16);       // K = kk+16 .. kk+23
#pragma unroll
    for (int mt = 0; mt < M_TILES; ++mt) {
      const unsigned short* arow = As + (mt * 16 + m) * K3;
      union { uint4 q[2]; v16bf v; } a;
      a.q[0] = *(const uint4*)(arow + kk);
      a.q[1] = *(const uint4*)(arow + kk + 16);
      acc[mt] = __builtin_amdgcn_wmma_f32_16x16x32_bf16(
          /*neg_a=*/false, a.v, /*neg_b=*/false, b.v,
          /*c_mod=*/(short)0, acc[mt], /*reuse_a=*/false, /*reuse_b=*/false);
    }
  }

  // fused ReLU + store + per-channel BN partial sums (one atomic pair per lane)
  float s1 = 0.f, s2 = 0.f;
  const long long m0 = (long long)blockIdx.x * M_BLK;
#pragma unroll
  for (int mt = 0; mt < M_TILES; ++mt) {
#pragma unroll
    for (int r = 0; r < 8; ++r) {
      float v = acc[mt][r];
      v = v > 0.f ? v : 0.f;
      Y[(m0 + mt * 16 + r + 8 * g) * OUT_DIM + n] = v;
      s1 += v;
      s2 += v * v;
    }
  }
  atomicAdd(&bnsum[n], s1);
  atomicAdd(&bnsumsq[n], s2);
}

// -------------------------------------------------------------- BN finalize/apply
__global__ void bn_final(const float* __restrict__ bnsum, const float* __restrict__ bnsumsq,
                         const float* __restrict__ gamma, const float* __restrict__ beta,
                         float* __restrict__ scale, float* __restrict__ shift) {
  int c = threadIdx.x;
  if (c >= OUT_DIM) return;
  const float inv_n = 1.0f / (float)N_NODES;
  float mu  = bnsum[c] * inv_n;
  float var = bnsumsq[c] * inv_n - mu * mu;
  float sc  = gamma[c] * rsqrtf(var + BN_EPS);
  scale[c] = sc;
  shift[c] = beta[c] - mu * sc;
}

__global__ void bn_apply(const float4* __restrict__ Y, const float* __restrict__ scale,
                         const float* __restrict__ shift, float4* __restrict__ H) {
  long long idx = (long long)blockIdx.x * blockDim.x + threadIdx.x;
  if (idx >= (long long)N_NODES * (OUT_DIM / 4)) return;
  int c = (int)(idx & (OUT_DIM / 4 - 1)) * 4;
  float4 y = Y[idx];
  float4 r;
  r.x = y.x * scale[c + 0] + shift[c + 0];
  r.y = y.y * scale[c + 1] + shift[c + 1];
  r.z = y.z * scale[c + 2] + shift[c + 2];
  r.w = y.w * scale[c + 3] + shift[c + 3];
  H[idx] = r;
}

// ----------------------------------------------------------------- global pooling
__global__ void pool_acc(const float* __restrict__ h, const int* __restrict__ batch,
                         float* __restrict__ pool, float* __restrict__ cnt) {
  long long idx = (long long)blockIdx.x * blockDim.x + threadIdx.x;
  if (idx >= (long long)N_NODES * (OUT_DIM / 4)) return;
  int i = (int)(idx >> 5);
  int j = (int)(idx & 31) * 4;
  int b = batch[i];
  const float4 v = *(const float4*)(h + (long long)i * OUT_DIM + j);
  float* p = pool + b * OUT_DIM + j;
  atomicAdd(p + 0, v.x); atomicAdd(p + 1, v.y);
  atomicAdd(p + 2, v.z); atomicAdd(p + 3, v.w);
  if (j == 0) atomicAdd(&cnt[b], 1.0f);
}

__global__ void pool_div(const float* __restrict__ pool, const float* __restrict__ cnt,
                         float* __restrict__ out) {
  int idx = blockIdx.x * blockDim.x + threadIdx.x;
  if (idx >= NUM_GRAPHS * OUT_DIM) return;
  out[idx] = pool[idx] / fmaxf(cnt[idx >> 7], 1.0f);
}

// =============================================================== host-side launch
extern "C" void kernel_launch(void* const* d_in, const int* in_sizes, int n_in,
                              void* d_out, int out_size, void* d_ws, size_t ws_size,
                              hipStream_t stream) {
  const float* x    = (const float*)d_in[0];
  const int*  ei    = (const int*)d_in[1];
  const int*  batch = (const int*)d_in[2];
  const int E = in_sizes[1] / 2;
  const int* srcp = ei;
  const int* dstp = ei + E;

  const float *Ws[3], *Wi[3], *Wo[3], *gam[3], *bet[3];
  for (int l = 0; l < 3; ++l) {
    Ws[l]  = (const float*)d_in[3 + 5 * l];
    Wi[l]  = (const float*)d_in[4 + 5 * l];
    Wo[l]  = (const float*)d_in[5 + 5 * l];
    gam[l] = (const float*)d_in[6 + 5 * l];
    bet[l] = (const float*)d_in[7 + 5 * l];
  }

  // ---- workspace carve-out (256B aligned)
  size_t off = 0;
  auto carve = [&](size_t bytes) -> void* {
    void* p = (char*)d_ws + off;
    off = (off + bytes + 255) & ~(size_t)255;
    return p;
  };
  float*          h       = (float*)carve((size_t)N_NODES * OUT_DIM * 4);
  float*          Yb      = (float*)carve((size_t)N_NODES * OUT_DIM * 4);
  float*          agg_in  = (float*)carve((size_t)N_NODES * OUT_DIM * 4);
  float*          agg_out = (float*)carve((size_t)N_NODES * OUT_DIM * 4);
  unsigned short* Xcat    = (unsigned short*)carve((size_t)N_NODES * 384 * 2);
  unsigned short* Wt[3];
  Wt[0] = (unsigned short*)carve((size_t)OUT_DIM * 192 * 2);
  Wt[1] = (unsigned short*)carve((size_t)OUT_DIM * 384 * 2);
  Wt[2] = (unsigned short*)carve((size_t)OUT_DIM * 384 * 2);
  float* stats = (float*)carve(256 * 4);           // [sum(128) | sumsq(128)]
  float* bnsum = stats, *bnsumsq = stats + 128;
  float* scale = (float*)carve(OUT_DIM * 4);
  float* shift = (float*)carve(OUT_DIM * 4);
  float* poolb = (float*)carve((NUM_GRAPHS * OUT_DIM + NUM_GRAPHS) * 4);
  float* cnt   = poolb + NUM_GRAPHS * OUT_DIM;

  // ---- weight prep (bf16, transposed/stacked)
  for (int l = 0; l < 3; ++l) {
    int din = (l == 0) ? 64 : 128;
    int tot = OUT_DIM * 3 * din;
    wprep<<<(tot + 255) / 256, 256, 0, stream>>>(Ws[l], Wi[l], Wo[l], Wt[l], din);
  }

  // ---- layers
  for (int l = 0; l < 3; ++l) {
    const int din = (l == 0) ? 64 : 128;
    const int K3  = 3 * din;
    const float* hin = (l == 0) ? x : h;

    zero_f32<<<1024, 256, 0, stream>>>(agg_in,  (long long)N_NODES * din);
    zero_f32<<<1024, 256, 0, stream>>>(agg_out, (long long)N_NODES * din);
    zero_f32<<<1, 256, 0, stream>>>(stats, 256);

    long long sthreads = (long long)E * (din / 4);
    scatter_edges<<<(unsigned)((sthreads + 255) / 256), 256, 0, stream>>>(
        hin, srcp, dstp, agg_in, agg_out, E, din);

    long long pthreads = (long long)N_NODES * (K3 / 2);
    pack_bf16<<<(unsigned)((pthreads + 255) / 256), 256, 0, stream>>>(
        hin, agg_in, agg_out, (unsigned int*)Xcat, din);

    gemm_wmma<<<N_NODES / M_BLK, 256, 0, stream>>>(Xcat, Wt[l], Yb, bnsum, bnsumsq, K3);

    bn_final<<<1, 128, 0, stream>>>(bnsum, bnsumsq, gam[l], bet[l], scale, shift);

    long long athreads = (long long)N_NODES * (OUT_DIM / 4);
    bn_apply<<<(unsigned)((athreads + 255) / 256), 256, 0, stream>>>(
        (const float4*)Yb, scale, shift, (float4*)h);
  }

  // ---- global mean pool
  zero_f32<<<64, 256, 0, stream>>>(poolb, NUM_GRAPHS * OUT_DIM + NUM_GRAPHS);
  long long pt = (long long)N_NODES * (OUT_DIM / 4);
  pool_acc<<<(unsigned)((pt + 255) / 256), 256, 0, stream>>>(h, batch, poolb, cnt);
  pool_div<<<(NUM_GRAPHS * OUT_DIM + 255) / 256, 256, 0, stream>>>(poolb, cnt, (float*)d_out);
}
